// Gemma3Attention_24343874633861
// MI455X (gfx1250) — matrix operand order repaired and checked by reading, not compiled, not run on hardware
//
#include <hip/hip_runtime.h>
#include <hip/hip_bf16.h>
#include <cmath>

// ---------------- problem constants ----------------
#define B_    2
#define S_    4096
#define HID_  2560
#define H_    8
#define KV_   4
#define D_    256
#define SW_   1024
#define SCALE_ 0.0625f   // 1/sqrt(256)

// ---------------- vector types ----------------
typedef __attribute__((ext_vector_type(16))) __bf16 v16bf;
typedef __attribute__((ext_vector_type(8)))  __bf16 v8bf;
typedef __attribute__((ext_vector_type(8)))  float  v8f;

static __device__ __forceinline__ __bf16 f2bf(float x) {
  __hip_bfloat16 h = __float2bfloat16(x);
  return *reinterpret_cast<__bf16*>(&h);
}

// Build a 16-element bf16 fragment from two 16B-aligned 8-element chunks.
static __device__ __forceinline__ v16bf frag16(const __bf16* p0, const __bf16* p1) {
  v8bf a = *(const v8bf*)p0;
  v8bf b = *(const v8bf*)p1;
  return __builtin_shufflevector(a, b, 0,1,2,3,4,5,6,7,8,9,10,11,12,13,14,15);
}
// Fragment carried as v8f (bit-identical) so it can be pinned with "+v" asm fences.
static __device__ __forceinline__ v8f fragraw(const __bf16* p0, const __bf16* p1) {
  return __builtin_bit_cast(v8f, frag16(p0, p1));
}
// Scheduling fence: force a group of fragment loads to complete (one dscnt wait)
// before the consuming WMMAs issue, so WMMAs go back-to-back.
#define FENCE4(b) asm volatile("" : "+v"((b)[0]), "+v"((b)[1]), "+v"((b)[2]), "+v"((b)[3]))

// ---------------- CDNA5 async copy global->LDS (ASYNCcnt path) ----------------
// Generic LDS addresses carry the LDS byte offset in bits [31:0] (ISA 10.2 aperture).
static __device__ __forceinline__ void async_copy_b128(const void* lds_dst, const void* gsrc) {
  unsigned lo = (unsigned)(unsigned long long)(size_t)lds_dst;
  unsigned long long ga = (unsigned long long)(size_t)gsrc;
  asm volatile("global_load_async_to_lds_b128 %0, %1, off"
               :: "v"(lo), "v"(ga) : "memory");
}
static __device__ __forceinline__ void wait_async0() {
  asm volatile("s_wait_asynccnt 0" ::: "memory");
}

// ---------------- elementwise f32 -> bf16 ----------------
__global__ void cast_f32_bf16(const float* __restrict__ in, __bf16* __restrict__ out, long n) {
  long i = ((long)blockIdx.x * blockDim.x + threadIdx.x) * 4;
  if (i + 4 <= n) {
    float4 v = *(const float4*)(in + i);
    out[i + 0] = f2bf(v.x);
    out[i + 1] = f2bf(v.y);
    out[i + 2] = f2bf(v.z);
    out[i + 3] = f2bf(v.w);
  } else {
    for (; i < n; ++i) out[i] = f2bf(in[i]);
  }
}

// ---------------- transpose + cast: W[K,N] f32 -> Wt[N,K] bf16 ----------------
__global__ void transpose_cast_kernel(const float* __restrict__ in,
                                      __bf16* __restrict__ out, int K, int N) {
  __shared__ float tile[32][33];
  const int k0 = blockIdx.x * 32;
  const int n0 = blockIdx.y * 32;
  const int tx = threadIdx.x & 31;
  const int ty = threadIdx.x >> 5;   // 0..7
#pragma unroll
  for (int r = ty; r < 32; r += 8)
    tile[r][tx] = in[(size_t)(k0 + r) * N + n0 + tx];
  __syncthreads();
#pragma unroll
  for (int r = ty; r < 32; r += 8)
    out[(size_t)(n0 + r) * K + k0 + tx] = f2bf(tile[tx][r]);
}

// ---------------- bf16 WMMA GEMM: C[M,N] = A[M,K] * Bt[N,K]^T (f32 out) ----------------
// Block: 256 threads = 8 waves. Macro tile 128(M) x 128(N), K step 32.
// Bt tile double-buffered in LDS via async-to-LDS copies; rows padded to 40 elems.
// launch_bounds(.,1): allow full VGPR budget (no accumulator spills); latency is
// hidden in-wave by batched loads + async prefetch, not by occupancy.
#define BSTRIDE 40
__global__ void __launch_bounds__(256, 1)
gemm_bf16_wmma(const __bf16* __restrict__ A,    // [M,K]
               const __bf16* __restrict__ Bt,   // [N,K]
               float* __restrict__ C,
               int M, int N, int K) {
  __shared__ __align__(32) __bf16 Bs[2][128 * BSTRIDE];   // ~20 KB
  const int tid  = threadIdx.x;
  const int lane = tid & 31;
  const int wid  = tid >> 5;
  const int grp  = lane >> 4;
  const int l16  = lane & 15;
  const int m0   = blockIdx.x * 128 + wid * 16;
  const int n0   = blockIdx.y * 128;

  // staging role: thread t copies 32 bytes of row (n0 + nr), k-chunk kc
  const int nr = tid >> 1;           // 0..127
  const int kc = (tid & 1) * 16;     // 0 or 16

  v8f acc[8] = {};

  // prefetch first tile into buffer 0
  {
    const __bf16* g = Bt + (size_t)(n0 + nr) * K + kc;
    async_copy_b128(&Bs[0][nr * BSTRIDE + kc], g);
    async_copy_b128(&Bs[0][nr * BSTRIDE + kc + 8], g + 8);
  }
  wait_async0();
  __syncthreads();

  int buf = 0;
  for (int kk = 0; kk < K; kk += 32) {
    // prefetch next K tile into the other buffer
    if (kk + 32 < K) {
      const __bf16* g = Bt + (size_t)(n0 + nr) * K + kk + 32 + kc;
      async_copy_b128(&Bs[buf ^ 1][nr * BSTRIDE + kc], g);
      async_copy_b128(&Bs[buf ^ 1][nr * BSTRIDE + kc + 8], g + 8);
    }

    // A fragment: lane row = l16, K chunks grp*8 and 16+grp*8
    const __bf16* ap = A + (size_t)(m0 + l16) * K + kk + grp * 8;
    v16bf afrag = frag16(ap, ap + 16);

    // two fenced groups of 4 fragments: one dscnt wait per 4 WMMAs,
    // bounded fragment liveness (32 VGPRs per group)
#pragma unroll
    for (int gq = 0; gq < 2; ++gq) {
      v8f braw[4];
#pragma unroll
      for (int q = 0; q < 4; ++q) {
        const __bf16* bp = &Bs[buf][((gq * 4 + q) * 16 + l16) * BSTRIDE + grp * 8];
        braw[q] = fragraw(bp, bp + 16);
      }
      FENCE4(braw);
#pragma unroll
      for (int q = 0; q < 4; ++q)
        acc[gq * 4 + q] = __builtin_amdgcn_wmma_f32_16x16x32_bf16(
            false, afrag, false, __builtin_bit_cast(v16bf, braw[q]),
            (short)0, acc[gq * 4 + q], false, false);
    }

    wait_async0();
    __syncthreads();
    buf ^= 1;
  }

#pragma unroll
  for (int nt = 0; nt < 8; ++nt) {
#pragma unroll
    for (int r = 0; r < 8; ++r) {
      int m = m0 + r + 8 * grp;
      C[(size_t)m * N + n0 + nt * 16 + l16] = acc[nt][r];
    }
  }
}

// ---------------- RMSNorm + RoPE, [B,S,Hn,D] f32 -> [B,Hn,S,D] bf16 ----------------
__global__ void rmsnorm_rope_kernel(const float* __restrict__ x,
                                    const float* __restrict__ cosb,
                                    const float* __restrict__ sinb,
                                    const float* __restrict__ w,
                                    __bf16* __restrict__ out,
                                    int Hn) {
  __shared__ float red[256];
  __shared__ float xs[256];
  const int d = threadIdx.x;
  const int s = blockIdx.x;
  const int h = blockIdx.y;
  const int b = blockIdx.z;

  const size_t iidx = (((size_t)b * S_ + s) * Hn + h) * D_ + d;
  float v = x[iidx];
  red[d] = v * v;
  __syncthreads();
  for (int st = 128; st > 0; st >>= 1) {
    if (d < st) red[d] += red[d + st];
    __syncthreads();
  }
  float mean = red[0] * (1.0f / D_);
  float inv  = rsqrtf(mean + 1e-6f);
  float xn   = v * inv * (1.0f + w[d]);
  xs[d] = xn;
  __syncthreads();
  float rot = (d < D_ / 2) ? -xs[d + D_ / 2] : xs[d - D_ / 2];
  size_t cidx = ((size_t)b * S_ + s) * D_ + d;
  float o = xn * cosb[cidx] + rot * sinb[cidx];
  out[(((size_t)b * Hn + h) * S_ + s) * D_ + d] = f2bf(o);
}

// ---------------- V: [B,S,KV,D] f32 -> [B,KV,S,D] bf16 ----------------
__global__ void v_cast_kernel(const float* __restrict__ x, __bf16* __restrict__ out) {
  const int d  = threadIdx.x;
  const int s  = blockIdx.x;
  const int kv = blockIdx.y;
  const int b  = blockIdx.z;
  out[(((size_t)b * KV_ + kv) * S_ + s) * D_ + d] =
      f2bf(x[(((size_t)b * S_ + s) * KV_ + kv) * D_ + d]);
}

// ---------------- sliding-window flash attention ----------------
// Grid: (S/128, H, B). Block: 256 threads = 8 waves; wave w handles 16 queries.
// K/V tiles double-buffered: Kt via async-to-LDS, Vt via manual transposed stores.
#define KSTRIDE 264
#define VSTRIDE 40
#define PSTRIDE 40
__global__ void __launch_bounds__(256, 1)
flash_swa_kernel(const __bf16* __restrict__ qg,   // [B,H,S,D]
                 const __bf16* __restrict__ kg,   // [B,KV,S,D]
                 const __bf16* __restrict__ vg,   // [B,KV,S,D]
                 __bf16* __restrict__ attn) {     // [B,S,H,D]
  __shared__ __align__(32) __bf16 Kt[2][32 * KSTRIDE];     // ~33 KB
  __shared__ __align__(32) __bf16 Vt[2][256 * VSTRIDE];    // ~41 KB
  __shared__ __align__(32) __bf16 Pb[8 * 16 * PSTRIDE];    // ~10 KB

  const int tid  = threadIdx.x;
  const int lane = tid & 31;
  const int wid  = tid >> 5;
  const int grp  = lane >> 4;
  const int l16  = lane & 15;

  const int h   = blockIdx.y;
  const int b   = blockIdx.z;
  const int kvh = h / (H_ / KV_);
  const int qb  = blockIdx.x * 128;
  const int qwb = qb + wid * 16;

  // Q fragments (A-matrix 16x32 layout), 8 K-chunks over D=256
  v16bf qf[8];
  {
    const __bf16* qrow = qg + (((size_t)b * H_ + h) * S_ + qwb + l16) * D_;
#pragma unroll
    for (int kfi = 0; kfi < 8; ++kfi)
      qf[kfi] = frag16(qrow + kfi * 32 + grp * 8, qrow + kfi * 32 + 16 + grp * 8);
  }

  v8f   oacc[16] = {};
  float mrow[8], lrow[8];
#pragma unroll
  for (int r = 0; r < 8; ++r) { mrow[r] = -1e30f; lrow[r] = 0.0f; }

  const int klo = (qb - SW_ > 0) ? (qb - SW_) : 0;   // multiple of 32
  const int khi = qb + 127;

  const __bf16* kbase = kg + ((size_t)b * KV_ + kvh) * (size_t)S_ * D_;
  const __bf16* vbase = vg + ((size_t)b * KV_ + kvh) * (size_t)S_ * D_;

  const int kr = tid >> 3;           // staging: key row 0..31
  const int dc = (tid & 7) * 32;     // staging: d chunk

  // stage tile starting at key k0 into buffer bufi
  auto stage = [&](int k0, int bufi) {
    const __bf16* krow = kbase + (size_t)(k0 + kr) * D_ + dc;
    const __bf16* vrow = vbase + (size_t)(k0 + kr) * D_ + dc;
#pragma unroll
    for (int c4 = 0; c4 < 4; ++c4)
      async_copy_b128(&Kt[bufi][kr * KSTRIDE + dc + c4 * 8], krow + c4 * 8);
    v8bf vv[4];
#pragma unroll
    for (int c4 = 0; c4 < 4; ++c4) vv[c4] = *(const v8bf*)(vrow + c4 * 8);
#pragma unroll
    for (int c4 = 0; c4 < 4; ++c4)
#pragma unroll
      for (int j = 0; j < 8; ++j)
        Vt[bufi][(dc + c4 * 8 + j) * VSTRIDE + kr] = vv[c4][j];
  };

  stage(klo, 0);
  wait_async0();
  __syncthreads();

  int buf = 0;
  for (int k0 = klo; k0 <= khi; k0 += 32) {
    // prefetch next K/V tile while computing this one
    if (k0 + 32 <= khi) stage(k0 + 32, buf ^ 1);

    // ---- scores S = Q * K^T (16 queries x 32 keys); fragment loads batched ----
    float pv[2][8];
#pragma unroll
    for (int nt = 0; nt < 2; ++nt) {
      v8f sacc = {};
#pragma unroll
      for (int half = 0; half < 2; ++half) {
        v8f braw[4];
#pragma unroll
        for (int q = 0; q < 4; ++q) {
          const __bf16* bp =
              &Kt[buf][(nt * 16 + l16) * KSTRIDE + (half * 4 + q) * 32 + grp * 8];
          braw[q] = fragraw(bp, bp + 16);
        }
        FENCE4(braw);
#pragma unroll
        for (int q = 0; q < 4; ++q)
          sacc = __builtin_amdgcn_wmma_f32_16x16x32_bf16(
              false, qf[half * 4 + q], false, __builtin_bit_cast(v16bf, braw[q]),
              (short)0, sacc, false, false);
      }
#pragma unroll
      for (int r = 0; r < 8; ++r) {
        int i = qwb + r + 8 * grp;
        int j = k0 + nt * 16 + l16;
        float val = sacc[r] * SCALE_;
        pv[nt][r] = ((j <= i) && (i - j <= SW_)) ? val : -1e30f;
      }
    }

    // ---- online softmax (row reductions across 16 lanes) ----
    float alpha[8];
#pragma unroll
    for (int r = 0; r < 8; ++r) {
      float mx = fmaxf(pv[0][r], pv[1][r]);
#pragma unroll
      for (int sh = 1; sh < 16; sh <<= 1) mx = fmaxf(mx, __shfl_xor(mx, sh, 32));
      float nm = fmaxf(mrow[r], mx);
      alpha[r] = __expf(mrow[r] - nm);
      float p0 = (pv[0][r] <= -1e29f) ? 0.0f : __expf(pv[0][r] - nm);
      float p1 = (pv[1][r] <= -1e29f) ? 0.0f : __expf(pv[1][r] - nm);
      pv[0][r] = p0; pv[1][r] = p1;
      float ps = p0 + p1;
#pragma unroll
      for (int sh = 1; sh < 16; sh <<= 1) ps += __shfl_xor(ps, sh, 32);
      lrow[r] = lrow[r] * alpha[r] + ps;
      mrow[r] = nm;
    }

    // ---- P (C-layout) -> per-wave LDS -> A-fragment; rescale O; O += P*V ----
#pragma unroll
    for (int r = 0; r < 8; ++r) {
      int m = r + 8 * grp;
      Pb[wid * 16 * PSTRIDE + m * PSTRIDE + l16]      = f2bf(pv[0][r]);
      Pb[wid * 16 * PSTRIDE + m * PSTRIDE + 16 + l16] = f2bf(pv[1][r]);
    }
#pragma unroll
    for (int nt = 0; nt < 16; ++nt)
#pragma unroll
      for (int r = 0; r < 8; ++r) oacc[nt][r] *= alpha[r];

    v16bf pf = frag16(&Pb[wid * 16 * PSTRIDE + l16 * PSTRIDE + grp * 8],
                      &Pb[wid * 16 * PSTRIDE + l16 * PSTRIDE + 16 + grp * 8]);
#pragma unroll
    for (int g = 0; g < 4; ++g) {
      v8f braw[4];
#pragma unroll
      for (int q = 0; q < 4; ++q) {
        const __bf16* bp = &Vt[buf][((g * 4 + q) * 16 + l16) * VSTRIDE + grp * 8];
        braw[q] = fragraw(bp, bp + 16);
      }
      FENCE4(braw);
#pragma unroll
      for (int q = 0; q < 4; ++q)
        oacc[g * 4 + q] = __builtin_amdgcn_wmma_f32_16x16x32_bf16(
            false, pf, false, __builtin_bit_cast(v16bf, braw[q]),
            (short)0, oacc[g * 4 + q], false, false);
    }

    wait_async0();
    __syncthreads();
    buf ^= 1;
  }

  // ---- normalize and write attn out [B,S,H,D] bf16 ----
#pragma unroll
  for (int nt = 0; nt < 16; ++nt) {
#pragma unroll
    for (int r = 0; r < 8; ++r) {
      int i = qwb + r + 8 * grp;
      attn[(((size_t)b * S_ + i) * H_ + h) * D_ + nt * 16 + l16] =
          f2bf(oacc[nt][r] / lrow[r]);
    }
  }
}

// ---------------- host launcher ----------------
extern "C" void kernel_launch(void* const* d_in, const int* in_sizes, int n_in,
                              void* d_out, int out_size, void* d_ws, size_t ws_size,
                              hipStream_t stream) {
  const float* hs   = (const float*)d_in[0];
  const float* cosb = (const float*)d_in[1];
  const float* sinb = (const float*)d_in[2];
  const float* Wq   = (const float*)d_in[3];
  const float* Wk   = (const float*)d_in[4];
  const float* Wv   = (const float*)d_in[5];
  const float* Wo   = (const float*)d_in[6];
  const float* qnw  = (const float*)d_in[7];
  const float* knw  = (const float*)d_in[8];
  float* out = (float*)d_out;

  const size_t M   = (size_t)B_ * S_;     // 8192
  const size_t HD  = (size_t)H_ * D_;     // 2048
  const size_t KVD = (size_t)KV_ * D_;    // 1024

  char* wp = (char*)d_ws;
  auto alloc = [&](size_t bytes) {
    char* p = wp;
    wp += (bytes + 255) & ~(size_t)255;
    return p;
  };
  __bf16* hs_b  = (__bf16*)alloc(M * HID_ * 2);
  __bf16* wqT   = (__bf16*)alloc((size_t)HID_ * HD * 2);   // [HD, HID]
  __bf16* wkT   = (__bf16*)alloc((size_t)HID_ * KVD * 2);  // [KVD, HID]
  __bf16* wvT   = (__bf16*)alloc((size_t)HID_ * KVD * 2);  // [KVD, HID]
  __bf16* woT   = (__bf16*)alloc(HD * HID_ * 2);           // [HID, HD]
  float*  qf32  = (float*)alloc(M * HD * 4);
  float*  kf32  = (float*)alloc(M * KVD * 4);
  float*  vf32  = (float*)alloc(M * KVD * 4);
  __bf16* qb16  = (__bf16*)alloc(M * HD * 2);    // [B,H,S,D]
  __bf16* kb16  = (__bf16*)alloc(M * KVD * 2);   // [B,KV,S,D]
  __bf16* vb16  = (__bf16*)alloc(M * KVD * 2);   // [B,KV,S,D]
  __bf16* attnb = (__bf16*)alloc(M * HD * 2);    // [B,S,H,D]

  // hidden states: straight cast
  {
    long n = (long)M * HID_;
    cast_f32_bf16<<<dim3((unsigned)((n / 4 + 255) / 256)), dim3(256), 0, stream>>>(
        hs, hs_b, n);
  }
  // weights: transpose + cast (so GEMM B-tiles are straight async copies)
  transpose_cast_kernel<<<dim3(HID_ / 32, HD / 32), dim3(256), 0, stream>>>(
      Wq, wqT, HID_, (int)HD);
  transpose_cast_kernel<<<dim3(HID_ / 32, KVD / 32), dim3(256), 0, stream>>>(
      Wk, wkT, HID_, (int)KVD);
  transpose_cast_kernel<<<dim3(HID_ / 32, KVD / 32), dim3(256), 0, stream>>>(
      Wv, wvT, HID_, (int)KVD);
  transpose_cast_kernel<<<dim3(HD / 32, HID_ / 32), dim3(256), 0, stream>>>(
      Wo, woT, (int)HD, HID_);

  // QKV projections (N tiles of 128)
  gemm_bf16_wmma<<<dim3(M / 128, HD / 128), dim3(256), 0, stream>>>(
      hs_b, wqT, qf32, (int)M, (int)HD, HID_);
  gemm_bf16_wmma<<<dim3(M / 128, KVD / 128), dim3(256), 0, stream>>>(
      hs_b, wkT, kf32, (int)M, (int)KVD, HID_);
  gemm_bf16_wmma<<<dim3(M / 128, KVD / 128), dim3(256), 0, stream>>>(
      hs_b, wvT, vf32, (int)M, (int)KVD, HID_);

  // RMSNorm + RoPE (q, k); transpose-cast v
  rmsnorm_rope_kernel<<<dim3(S_, H_, B_), dim3(256), 0, stream>>>(
      qf32, cosb, sinb, qnw, qb16, H_);
  rmsnorm_rope_kernel<<<dim3(S_, KV_, B_), dim3(256), 0, stream>>>(
      kf32, cosb, sinb, knw, kb16, KV_);
  v_cast_kernel<<<dim3(S_, KV_, B_), dim3(256), 0, stream>>>(vf32, vb16);

  // sliding-window flash attention
  flash_swa_kernel<<<dim3(S_ / 128, H_, B_), dim3(256), 0, stream>>>(
      qb16, kb16, vb16, attnb);

  // output projection -> f32 d_out
  gemm_bf16_wmma<<<dim3(M / 128, HID_ / 128), dim3(256), 0, stream>>>(
      attnb, woT, out, (int)M, HID_, (int)HD);
}
